// GAT_36850819400290
// MI455X (gfx1250) — matrix operand order repaired and checked by reading, not compiled, not run on hardware
//
#include <hip/hip_runtime.h>
#include <hip/hip_bf16.h>

// ---------------------------------------------------------------------------
// GAT (B=8, N=2048, D=128, L=3), flash-style masked softmax attention.
// bf16 WMMA (v_wmma_f32_16x16x32_bf16) for both GEMMs, fp32 softmax/residual.
// All bf16 conversion + fragment permutation is done in cheap prep kernels;
// hot loops use contiguous b128 fragment loads only.
// ---------------------------------------------------------------------------

typedef __attribute__((ext_vector_type(16))) __bf16 v16bf;
typedef __attribute__((ext_vector_type(8)))  float  v8f;

constexpr int   kB = 8, kN = 2048, kD = 128, kL = 3;
constexpr float kAlpha  = 0.2f;
constexpr float kNegInf = -9e15f;

// fp32 -> bf16 round-to-nearest-even (prep kernels / producers only)
__device__ __forceinline__ unsigned short f2bfbits(float f) {
    unsigned u = __builtin_bit_cast(unsigned, f);
    unsigned r = u + 0x7FFFu + ((u >> 16) & 1u);
    return (unsigned short)(r >> 16);
}
__device__ __forceinline__ __bf16 f2bf(float f) {
    unsigned short s = f2bfbits(f);
    return __builtin_bit_cast(__bf16, s);
}

// 16-bit A-matrix 16x32 K index for element e of v16bf, half-wave `group`
// (ISA 7.12.2: VGPR0..3 hold K=0..7 (+8 for lanes 16-31), VGPR4..7 K=16..23 (+8))
__device__ __forceinline__ int a_klocal(int e, int group) {
    int vg = e >> 1;
    int base = (vg < 4) ? (vg * 2) : (16 + (vg - 4) * 2);
    return base + (e & 1) + group * 8;
}
// inverse: given K (0..31) -> (group, element)
__device__ __forceinline__ void a_kinv(int k, int& g, int& e) {
    g = (k >> 3) & 1;
    int k0 = k - 8 * g;                    // {0..7} U {16..23}
    int vg = (k0 < 8) ? (k0 >> 1) : (4 + ((k0 - 16) >> 1));
    e = 2 * vg + (k0 & 1);
}

// ---------------------------------------------------------------------------
// Prep 0 (once per call): pack adj (int32 0/1) into bitmask words.
// One wave reads 32 consecutive ints, OR-reduces lane bits, lane0 stores word.
// ---------------------------------------------------------------------------
__global__ __launch_bounds__(256)
void gat_maskpack(const int* __restrict__ adj, unsigned* __restrict__ mask) {
    const size_t idx = (size_t)blockIdx.x * 256 + threadIdx.x;
    const int lane = threadIdx.x & 31;
    unsigned bit = (adj[idx] > 0) ? (1u << lane) : 0u;
    #pragma unroll
    for (int m = 16; m >= 1; m >>= 1)
        bit |= (unsigned)__shfl_xor((int)bit, m, 32);
    if (lane == 0) mask[idx >> 5] = bit;
}

// ---------------------------------------------------------------------------
// Prep 1 (per layer): x fp32 -> bf16 in A-fragment order xF[mt][kc][lane][e].
// ---------------------------------------------------------------------------
__global__ __launch_bounds__(256)
void gat_xcvt(const float* __restrict__ x, __bf16* __restrict__ xF) {
    const int o = blockIdx.x * 256 + threadIdx.x;       // 0 .. B*N*D-1
    const int e  = o & 15;
    const int l  = (o >> 4) & 31;
    const int kc = (o >> 9) & 3;
    const int mt = o >> 11;
    const int m = l & 15, g = l >> 4;
    const int k = kc * 32 + g * 16 + e;
    xF[o] = f2bf(x[((size_t)mt * 16 + m) * kD + k]);
}

// ---------------------------------------------------------------------------
// Prep 2 (per layer): W fp32 [k][n] -> WT bf16 [n][k]  (B-fragment friendly).
// ---------------------------------------------------------------------------
__global__ __launch_bounds__(256)
void gat_wcvt(const float* __restrict__ W, __bf16* __restrict__ WT) {
    const int o = blockIdx.x * 256 + threadIdx.x;       // 0 .. D*D-1
    const int n = o >> 7, k = o & 127;
    WT[o] = f2bf(W[(size_t)k * kD + n]);
}

// ---------------------------------------------------------------------------
// Kernel 1: h = relu(x @ W + b), written TRANSPOSED as hT[b][d][n] bf16.
// No LDS, no barriers: A-frag = 2xb128 from xF, B-frag = 2xb128 from WT,
// C rows are contiguous -> one packed b128 store per lane.
// grid = (B*N)/16 blocks of 256 (8 waves = 8 column tiles).
// ---------------------------------------------------------------------------
__global__ __launch_bounds__(256)
void gat_gemm1(const __bf16* __restrict__ xF, const __bf16* __restrict__ WT,
               const float* __restrict__ bias, __bf16* __restrict__ hT) {
    const int tid = threadIdx.x;
    const int wave = tid >> 5, lane = tid & 31;
    const int group = lane >> 4, lm = lane & 15;
    const int mt = blockIdx.x;                 // 16-row M tile
    const int n0 = wave * 16;

    const __bf16* aP = xF + ((size_t)mt * 128 + lane) * 16;       // +kc*512
    const __bf16* bP = WT + (size_t)(n0 + lm) * kD + group * 16;  // +kc*32

    v8f acc = {0.f, 0.f, 0.f, 0.f, 0.f, 0.f, 0.f, 0.f};
    #pragma unroll
    for (int kc = 0; kc < 4; ++kc) {
        v16bf a = *reinterpret_cast<const v16bf*>(aP + kc * 512);
        v16bf b = *reinterpret_cast<const v16bf*>(bP + kc * 32);
        acc = __builtin_amdgcn_wmma_f32_16x16x32_bf16(
            false, a, false, b, (short)0, acc, false, false);
    }

    const float bv = bias[n0 + lm];
    const int m0   = mt * 16;
    const int bb   = m0 >> 11;                  // batch  (m0 / N)
    const int rowb = (m0 & (kN - 1)) + group * 8;
    unsigned pk[4];
    #pragma unroll
    for (int rp = 0; rp < 4; ++rp) {            // rows r=2rp, 2rp+1 (M=r+8g)
        float h0 = fmaxf(acc[2 * rp]     + bv, 0.f);
        float h1 = fmaxf(acc[2 * rp + 1] + bv, 0.f);
        pk[rp] = (unsigned)f2bfbits(h0) | ((unsigned)f2bfbits(h1) << 16);
    }
    uint4 v = make_uint4(pk[0], pk[1], pk[2], pk[3]);
    *reinterpret_cast<uint4*>(hT + ((size_t)bb * kD + n0 + lm) * kN + rowb) = v;
}

// ---------------------------------------------------------------------------
// Kernel 2: s1 = h.a1, s2 = h.a2 from hT. Each thread owns 8 consecutive j,
// loops d with uint4 loads (8 bf16), mask/shift decode.  grid = B*N/(256*8).
// ---------------------------------------------------------------------------
__global__ __launch_bounds__(256)
void gat_s12(const __bf16* __restrict__ hT, const float* __restrict__ a1,
             const float* __restrict__ a2, float* __restrict__ s1,
             float* __restrict__ s2) {
    const int t = blockIdx.x * 256 + threadIdx.x;
    const int o = t * 8;                         // flat row start
    const int b = o >> 11;                       // / N
    const int j = o & (kN - 1);
    const __bf16* base = hT + (size_t)b * kD * kN + j;
    float p1[8], p2[8];
    #pragma unroll
    for (int i = 0; i < 8; ++i) { p1[i] = 0.f; p2[i] = 0.f; }
    for (int d = 0; d < kD; ++d) {
        uint4 v = *reinterpret_cast<const uint4*>(base + (size_t)d * kN);
        const float w1 = a1[d], w2 = a2[d];
        unsigned w[4] = {v.x, v.y, v.z, v.w};
        #pragma unroll
        for (int q = 0; q < 4; ++q) {
            float lo = __builtin_bit_cast(float, w[q] << 16);
            float hi = __builtin_bit_cast(float, w[q] & 0xFFFF0000u);
            p1[2 * q]     += lo * w1; p2[2 * q]     += lo * w2;
            p1[2 * q + 1] += hi * w1; p2[2 * q + 1] += hi * w2;
        }
    }
    #pragma unroll
    for (int i = 0; i < 8; ++i) { s1[o + i] = p1[i]; s2[o + i] = p2[i]; }
}

// ---------------------------------------------------------------------------
// Kernel 3: online-softmax attention + residual.
// Block = (batch b, 16-row i-tile); wave w owns d-cols [16w,16w+16).
// Per 32-wide j chunk: half-wave (w,g) scores row 2w+g using one broadcast
// mask word; P written to LDS in A-frag order (2xb128 read); B-frag read
// straight from global hT (L2-resident).  One WMMA per wave per chunk.
// ---------------------------------------------------------------------------
__global__ __launch_bounds__(256)
void gat_attn(const float* __restrict__ xin, const __bf16* __restrict__ hT,
              const float* __restrict__ s1, const float* __restrict__ s2,
              const unsigned* __restrict__ mask, float* __restrict__ xout) {
    __shared__ float s1s[16], m_row[16], l_row[16];
    __shared__ alignas(16) float scale_row[16];
    __shared__ alignas(16) __bf16 PbF[32][24];   // A-frag order, 48B stride

    const int tid = threadIdx.x;
    const int wave = tid >> 5, lane = tid & 31;
    const int group = lane >> 4, lm = lane & 15;
    const int b  = blockIdx.x >> 7;              // / (N/16)
    const int i0 = (blockIdx.x & 127) * 16;
    const size_t bN = (size_t)b * kN;

    if (tid < 16) {
        s1s[tid]   = s1[bN + i0 + tid];
        m_row[tid] = -__builtin_inff();
        l_row[tid] = 0.f;
    }
    __syncthreads();

    const int d0   = wave * 16;
    const int rloc = wave * 2 + group;           // row owned in score phase
    const unsigned* mrow = mask + (((bN + (size_t)(i0 + rloc)) * kN) >> 5);
    const __bf16* hTrow = hT + ((size_t)b * kD + d0 + lm) * kN + group * 16;

    v8f acc = {0.f, 0.f, 0.f, 0.f, 0.f, 0.f, 0.f, 0.f};

    for (int jc = 0; jc < kN / 32; ++jc) {
        const int j0 = jc * 32;
        __builtin_prefetch(hTrow + j0 + 64, 0, 1);   // two chunks ahead in hT

        // ---- scores for my row, online softmax update ----
        const float m_old = m_row[rloc];
        const float s1v = s1s[rloc];
        float e_a = s1v + s2[bN + j0 + lm];
        float e_b = s1v + s2[bN + j0 + lm + 16];
        e_a = e_a > 0.f ? e_a : kAlpha * e_a;        // leaky relu
        e_b = e_b > 0.f ? e_b : kAlpha * e_b;
        const unsigned mw = mrow[jc];                // broadcast word
        if (!((mw >> lm) & 1u))        e_a = kNegInf;
        if (!((mw >> (lm + 16)) & 1u)) e_b = kNegInf;

        float mc = fmaxf(e_a, e_b);
        #pragma unroll
        for (int m = 8; m >= 1; m >>= 1) mc = fmaxf(mc, __shfl_xor(mc, m, 16));
        const float m_new = fmaxf(m_old, mc);
        const float p_a = __expf(e_a - m_new);
        const float p_b = __expf(e_b - m_new);
        float sc = p_a + p_b;
        #pragma unroll
        for (int m = 8; m >= 1; m >>= 1) sc += __shfl_xor(sc, m, 16);
        const float scl = __expf(m_old - m_new);

        // scatter P into A-frag order: value (row=rloc, k) -> PbF[g*16+row][e]
        PbF[rloc][lm]      = f2bf(p_a);
        PbF[16 + rloc][lm] = f2bf(p_b);
        if (lm == 0) {
            m_row[rloc]     = m_new;
            l_row[rloc]     = l_row[rloc] * scl + sc;
            scale_row[rloc] = scl;
        }
        __syncthreads();

        // ---- rescale accumulator + P(16x32) @ H(32x16) via WMMA ----
        const float4* sv = reinterpret_cast<const float4*>(&scale_row[group * 8]);
        const float4 sA = sv[0], sB = sv[1];
        acc[0] *= sA.x; acc[1] *= sA.y; acc[2] *= sA.z; acc[3] *= sA.w;
        acc[4] *= sB.x; acc[5] *= sB.y; acc[6] *= sB.z; acc[7] *= sB.w;

        v16bf a  = *reinterpret_cast<const v16bf*>(&PbF[lane][0]);   // 2xb128 LDS
        v16bf bb = *reinterpret_cast<const v16bf*>(hTrow + j0);      // 2xb128 global
        acc = __builtin_amdgcn_wmma_f32_16x16x32_bf16(
            false, a, false, bb, (short)0, acc, false, false);
        __syncthreads();
    }

    // ---- residual: x + acc / l ----
    #pragma unroll
    for (int r = 0; r < 8; ++r) {
        const int rl = r + group * 8;
        const size_t idx = (bN + (size_t)(i0 + rl)) * kD + d0 + lm;
        xout[idx] = xin[idx] + acc[r] / l_row[rl];
    }
}

// ---------------------------------------------------------------------------
extern "C" void kernel_launch(void* const* d_in, const int* in_sizes, int n_in,
                              void* d_out, int out_size, void* d_ws, size_t ws_size,
                              hipStream_t stream) {
    (void)in_sizes; (void)n_in; (void)out_size; (void)ws_size;
    const float* x0     = (const float*)d_in[0];
    const int*   adj    = (const int*)  d_in[1];
    const float* Wg     = (const float*)d_in[2];
    const float* bg     = (const float*)d_in[3];
    const float* attn_a = (const float*)d_in[4];

    char* ws = (char*)d_ws;
    size_t off = 0;
    unsigned* mask = (unsigned*)(ws + off); off += (size_t)kB * kN * kN / 8;  // 4 MB
    __bf16*   hT   = (__bf16*)  (ws + off); off += (size_t)kB * kN * kD * 2;  // 4 MB
    __bf16*   xF   = (__bf16*)  (ws + off); off += (size_t)kB * kN * kD * 2;  // 4 MB
    __bf16*   WT   = (__bf16*)  (ws + off); off += (size_t)kD * kD * 2;       // 32 KB
    float*    s1   = (float*)   (ws + off); off += (size_t)kB * kN * 4;       // 64 KB
    float*    s2   = (float*)   (ws + off); off += (size_t)kB * kN * 4;       // 64 KB
    float*    xA   = (float*)   (ws + off);                                   // 8 MB

    const dim3 blk(256);
    gat_maskpack<<<dim3((kB * kN * kN) / 256), blk, 0, stream>>>(adj, mask);

    const float* xin = x0;
    for (int layer = 0; layer < kL; ++layer) {
        // layer0: d_in -> xA ; layer1: xA -> xA (in-place safe: each x element
        // is read only by its writer thread, after the j-loop) ; layer2: -> out
        float* xout = (layer == kL - 1) ? (float*)d_out : xA;

        gat_xcvt<<<dim3(kB * kN * kD / 256), blk, 0, stream>>>(xin, xF);
        gat_wcvt<<<dim3(kD * kD / 256), blk, 0, stream>>>(
            Wg + (size_t)layer * kD * kD, WT);
        gat_gemm1<<<dim3(kB * kN / 16), blk, 0, stream>>>(
            xF, WT, bg + (size_t)layer * kD, hT);
        gat_s12<<<dim3(kB * kN / (256 * 8)), blk, 0, stream>>>(
            hT, attn_a + (size_t)layer * 2 * kD,
            attn_a + (size_t)layer * 2 * kD + kD, s1, s2);
        gat_attn<<<dim3(kB * (kN / 16)), blk, 0, stream>>>(
            xin, hT, s1, s2, mask, xout);

        xin = xout;
    }
}
